// TGNModel_17592186044553
// MI455X (gfx1250) — matrix operand order repaired and checked by reading, not compiled, not run on hardware
//
#include <hip/hip_runtime.h>
#include <hip/hip_bf16.h>

#define Dm 100      // in channels
#define Tm 100      // time dim
#define Cc 50       // per-head channels
#define HC 100      // H*C
#define Mm 172      // msg dim
#define EDm 272     // edge dim
#define KNODE 128   // D padded to 4 k-tiles of 32
#define KEDGE 288   // ED padded to 9 k-tiles of 32
#define NODE_COLS 400 // q|k|v|skip fused output columns
#define WPAD 112    // HC padded to 7 n-tiles of 16
#define EW 2        // waves per block in edge kernel (keeps static LDS ~33KB)

typedef __attribute__((ext_vector_type(16))) __bf16 v16bf;
typedef __attribute__((ext_vector_type(8)))  float  v8f;

union FragBF { v16bf v; uint32_t u[8]; };

__device__ __forceinline__ v8f wmma_bf16(v16bf a, v16bf b, v8f c) {
  return __builtin_amdgcn_wmma_f32_16x16x32_bf16(false, a, false, b, (short)0, c, false, false);
}

__device__ __forceinline__ uint32_t pkbf(float a, float b) {
  union { __bf16 h[2]; uint32_t u; } r;
  r.h[0] = (__bf16)a; r.h[1] = (__bf16)b;
  return r.u;
}

// order-preserving float<->uint for atomic float max via atomicMax(u32)
__device__ __forceinline__ unsigned f2o(float f) {
  unsigned u = __float_as_uint(f);
  return (u & 0x80000000u) ? ~u : (u | 0x80000000u);
}
__device__ __forceinline__ float o2f(unsigned u) {
  u = (u & 0x80000000u) ? (u & 0x7FFFFFFFu) : ~u;
  return __uint_as_float(u);
}

__global__ void k_init(unsigned* amax, float* denom, int n2) {
  int i = blockIdx.x * blockDim.x + threadIdx.x;
  if (i < n2) { amax[i] = 0x007FFFFFu; /* f2o(-inf) */ denom[i] = 0.0f; }
}

__global__ void k_prep(const float* Wq, const float* bq, const float* Wk, const float* bk,
                       const float* Wv, const float* bv, const float* We,
                       const float* Ws, const float* bs,
                       __bf16* Wnode, float* ball, __bf16* Wep) {
  int tid = blockIdx.x * blockDim.x + threadIdx.x;
  int nth = gridDim.x * blockDim.x;
  // fused node weight [KNODE][NODE_COLS], K-padded with zeros
  for (int i = tid; i < KNODE * NODE_COLS; i += nth) {
    int kk = i / NODE_COLS, j = i % NODE_COLS;
    int which = j / HC, c = j % HC;
    float v = 0.f;
    if (kk < Dm) {
      const float* W = which == 0 ? Wq : which == 1 ? Wk : which == 2 ? Wv : Ws;
      v = W[kk * HC + c];
    }
    Wnode[i] = (__bf16)v;
  }
  for (int i = tid; i < NODE_COLS; i += nth) {
    int which = i / HC, c = i % HC;
    const float* b = which == 0 ? bq : which == 1 ? bk : which == 2 ? bv : bs;
    ball[i] = b[c];
  }
  // edge weight [KEDGE][WPAD], zero padded rows/cols
  for (int i = tid; i < KEDGE * WPAD; i += nth) {
    int kk = i / WPAD, n = i % WPAD;
    float v = (kk < EDm && n < HC) ? We[kk * HC + n] : 0.f;
    Wep[i] = (__bf16)v;
  }
}

// one wave computes a 16-row M-tile of x @ [Wq|Wk|Wv|Ws] (+bias): 25 n-tiles x 4 k-tiles
__global__ __launch_bounds__(256) void k_node(const float* __restrict__ x,
    const __bf16* __restrict__ Wnode, const float* __restrict__ ball,
    float* __restrict__ qb, float* __restrict__ kb, float* __restrict__ vb,
    float* __restrict__ outS, int n) {
  int lane = threadIdx.x & 31;
  int wid = blockIdx.x * (blockDim.x >> 5) + (threadIdx.x >> 5);
  int mtiles = (n + 15) >> 4;
  if (wid >= mtiles) return;               // wave-uniform: EXEC stays all-ones for WMMA
  int row = wid * 16 + (lane & 15);
  bool rvalid = row < n;
  const float* xr = x + (size_t)row * Dm;
  int h4 = (lane & 16) ? 4 : 0;
  FragBF a[4];
  #pragma unroll
  for (int kt = 0; kt < 4; ++kt) {
    #pragma unroll
    for (int vv = 0; vv < 8; ++vv) {
      int dw = kt * 16 + ((vv < 4) ? vv : vv + 4) + h4; // dword index = K/2 per ISA A-layout
      int K = dw * 2;
      a[kt].u[vv] = (rvalid && (K + 1) < Dm) ? pkbf(xr[K], xr[K + 1]) : 0u;
    }
  }
  // incremental per-lane (buffer, column) tracking: no div/mod in the hot loop
  int c = lane & 15;       // column within current destination buffer
  int nc = lane & 15;      // global fused column (bias index)
  int which = 0;           // 0=q 1=k 2=v 3=skip
  #pragma unroll 1
  for (int nt = 0; nt < 25; ++nt) {
    v8f acc = {0.f, 0.f, 0.f, 0.f, 0.f, 0.f, 0.f, 0.f};
    #pragma unroll
    for (int kt = 0; kt < 4; ++kt) {
      FragBF b;
      const __bf16* bp = Wnode + (size_t)(kt * 32) * NODE_COLS + nt * 16 + (lane & 15);
      #pragma unroll
      for (int i = 0; i < 8; ++i) {
        int K = 2 * (((i < 4) ? i : i + 4) + h4);
        b.v[2 * i] = bp[(size_t)K * NODE_COLS];
        b.v[2 * i + 1] = bp[(size_t)(K + 1) * NODE_COLS];
      }
      acc = wmma_bf16(a[kt].v, b.v, acc);
    }
    float* dstb = which == 0 ? qb : which == 1 ? kb : which == 2 ? vb : outS;
    float bias = ball[nc];
    #pragma unroll
    for (int r = 0; r < 8; ++r) {
      int m = wid * 16 + r + ((lane & 16) ? 8 : 0);
      if (m < n) dstb[(size_t)m * HC + c] = acc[r] + bias;
    }
    c += 16; nc += 16;
    if (c >= HC) { c -= HC; ++which; }
  }
}

// one wave per 16-edge tile: fused cos-encoding + edge GEMM + attention logits + (v[src]+e)
__global__ __launch_bounds__(EW * 32) void k_edge(
    const float* __restrict__ last_update, const float* __restrict__ tarr,
    const float* __restrict__ msg, const float* __restrict__ Wt, const float* __restrict__ bt,
    const int* __restrict__ ei, const __bf16* __restrict__ Wep,
    const float* __restrict__ qb, const float* __restrict__ kb, const float* __restrict__ vb,
    float* __restrict__ alpha, __bf16* __restrict__ wbuf, unsigned* __restrict__ amax, int nE) {
  __shared__ __align__(16) __bf16 sA[EW][16][KEDGE];
  __shared__ __align__(16) float  sE[EW][16][WPAD];
  __shared__ float sRel[EW][16];
  __shared__ int   sSrc[EW][16];
  __shared__ int   sDst[EW][16];

  int lane = threadIdx.x & 31;
  int w = threadIdx.x >> 5;
  int tile = blockIdx.x * EW + w;
  int e0 = tile * 16;

  if (lane < 16) {
    int e = e0 + lane;
    int s = 0, d = 0; float rt = 0.f;
    if (e < nE) { s = ei[e]; d = ei[nE + e]; rt = last_update[s] - tarr[e]; }
    sSrc[w][lane] = s; sDst[w][lane] = d; sRel[w][lane] = rt;
  }
  __syncthreads();

  // build 16x288 bf16 edge_attr tile: [cos(rel_t*Wt+bt) | msg | 0-pad]; lane-strided
  // columns keep msg loads coalesced, no div/mod
  #pragma unroll 1
  for (int r = 0; r < 16; ++r) {
    int e = e0 + r;
    bool ev = e < nE;
    float rt = sRel[w][r];
    const float* mrow = msg + (size_t)e * Mm;
    for (int cc = lane; cc < KEDGE; cc += 32) {
      float v = 0.f;
      if (ev) {
        if (cc < Tm)       v = cosf(rt * Wt[cc] + bt[cc]);
        else if (cc < EDm) v = mrow[cc - Tm];
      }
      sA[w][r][cc] = (__bf16)v;
    }
  }
  __syncthreads();

  // e = edge_attr @ We : kt-outer with 7 live accumulators => each A fragment read
  // from LDS once; 7 WMMAs + 112 strided bf16 B-loads per k-step form clauses
  int h4 = (lane & 16) ? 4 : 0;
  const uint32_t* arow = (const uint32_t*)&sA[w][lane & 15][0];
  v8f zero8 = {0.f, 0.f, 0.f, 0.f, 0.f, 0.f, 0.f, 0.f};
  v8f acc[7];
  #pragma unroll
  for (int nt = 0; nt < 7; ++nt) acc[nt] = zero8;
  #pragma unroll 1
  for (int kt = 0; kt < 9; ++kt) {
    FragBF a;
    #pragma unroll
    for (int vv = 0; vv < 8; ++vv)
      a.u[vv] = arow[kt * 16 + ((vv < 4) ? vv : vv + 4) + h4];
    const __bf16* bcol = Wep + (size_t)(kt * 32) * WPAD + (lane & 15);
    #pragma unroll
    for (int nt = 0; nt < 7; ++nt) {
      FragBF b;
      #pragma unroll
      for (int i = 0; i < 8; ++i) {
        int K = 2 * (((i < 4) ? i : i + 4) + h4);
        b.v[2 * i] = bcol[(size_t)K * WPAD + nt * 16];
        b.v[2 * i + 1] = bcol[(size_t)(K + 1) * WPAD + nt * 16];
      }
      acc[nt] = wmma_bf16(a.v, b.v, acc[nt]);
    }
  }
  #pragma unroll
  for (int nt = 0; nt < 7; ++nt) {
    int ccol = nt * 16 + (lane & 15);
    int m = (lane & 16) ? 8 : 0;
    #pragma unroll
    for (int r = 0; r < 8; ++r)
      sE[w][m + r][ccol] = acc[nt][r];
  }
  __syncthreads();

  // attention logits: lane <-> (edge, head); segment max via monotone-uint atomicMax
  {
    int le = lane >> 1, head = lane & 1;
    int e = e0 + le;
    if (e < nE) {
      int s = sSrc[w][le], d = sDst[w][le];
      const float2* q2 = (const float2*)(qb + (size_t)d * HC + head * Cc);
      const float2* k2 = (const float2*)(kb + (size_t)s * HC + head * Cc);
      const float* er = &sE[w][le][head * Cc];
      float dotv = 0.f;
      #pragma unroll 5
      for (int c2 = 0; c2 < Cc / 2; ++c2) {
        float2 qv = q2[c2], kv = k2[c2];
        dotv += qv.x * (kv.x + er[2 * c2]) + qv.y * (kv.y + er[2 * c2 + 1]);
      }
      dotv *= 0.14142135623730951f;   // 1/sqrt(50)
      alpha[(size_t)e * 2 + head] = dotv;
      atomicMax(&amax[(size_t)d * 2 + head], f2o(dotv));
    }
  }

  // stage w = v[src] + e as bf16 (reused by k_out)
  #pragma unroll 1
  for (int r = 0; r < 16; ++r) {
    int e = e0 + r;
    if (e >= nE) continue;   // wave-uniform
    const float* vr = vb + (size_t)sSrc[w][r] * HC;
    __bf16* wr = wbuf + (size_t)e * HC;
    for (int cc = lane; cc < HC; cc += 32)
      wr[cc] = (__bf16)(vr[cc] + sE[w][r][cc]);
  }
}

__global__ void k_exp(float* alpha, const unsigned* amax, float* denom,
                      const int* ei, int nE) {
  int i = blockIdx.x * blockDim.x + threadIdx.x;
  if (i >= nE * 2) return;
  int e = i >> 1, h = i & 1;
  int d = ei[nE + e];
  float m = o2f(amax[(size_t)d * 2 + h]);
  if (m < -1e37f) m = 0.f;                 // mirror reference's isfinite(amax) guard
  float ex = expf(alpha[i] - m);
  atomicAdd(&denom[(size_t)d * 2 + h], ex);
  alpha[i] = ex;
}

__global__ void k_out(const float* __restrict__ alpha, const float* __restrict__ denom,
                      const __bf16* __restrict__ wbuf, const int* __restrict__ ei,
                      float* out, int nE) {
  long long i = (long long)blockIdx.x * blockDim.x + threadIdx.x;
  if (i >= (long long)nE * HC) return;
  int e = (int)(i / HC), c = (int)(i % HC);
  int h = (c >= Cc) ? 1 : 0;
  int d = ei[nE + e];
  float attn = alpha[(size_t)e * 2 + h] / denom[(size_t)d * 2 + h];
  atomicAdd(&out[(size_t)d * HC + c], attn * (float)wbuf[(size_t)e * HC + c]);
}

extern "C" void kernel_launch(void* const* d_in, const int* in_sizes, int n_in,
                              void* d_out, int out_size, void* d_ws, size_t ws_size,
                              hipStream_t stream) {
  const float* x   = (const float*)d_in[0];
  const float* lu  = (const float*)d_in[1];
  const float* t   = (const float*)d_in[2];
  const float* msg = (const float*)d_in[3];
  const float* Wt  = (const float*)d_in[4];
  const float* bt  = (const float*)d_in[5];
  const float* Wq  = (const float*)d_in[6];
  const float* bq  = (const float*)d_in[7];
  const float* Wk  = (const float*)d_in[8];
  const float* bk  = (const float*)d_in[9];
  const float* Wv  = (const float*)d_in[10];
  const float* bv  = (const float*)d_in[11];
  const float* We  = (const float*)d_in[12];
  const float* Ws  = (const float*)d_in[13];
  const float* bs  = (const float*)d_in[14];
  const int* ei    = (const int*)d_in[15];
  int N = in_sizes[1];   // last_update is [N]
  int E = in_sizes[2];   // t is [E]
  float* out = (float*)d_out;

  char* ws = (char*)d_ws;
  size_t off = 0;
  auto take = [&](size_t bytes) -> char* {
    char* p = ws + off;
    off += (bytes + 255) & ~(size_t)255;
    return p;
  };
  unsigned* amax = (unsigned*)take((size_t)2 * N * 4);
  float* denom   = (float*)take((size_t)2 * N * 4);
  __bf16* Wnode  = (__bf16*)take((size_t)KNODE * NODE_COLS * 2);
  float* ball    = (float*)take((size_t)NODE_COLS * 4);
  __bf16* Wep    = (__bf16*)take((size_t)KEDGE * WPAD * 2);
  float* qb      = (float*)take((size_t)N * HC * 4);
  float* kb      = (float*)take((size_t)N * HC * 4);
  float* vb      = (float*)take((size_t)N * HC * 4);
  float* alpha   = (float*)take((size_t)E * 2 * 4);
  __bf16* wbuf   = (__bf16*)take((size_t)E * HC * 2);

  k_init<<<(2 * N + 255) / 256, 256, 0, stream>>>(amax, denom, 2 * N);
  k_prep<<<128, 256, 0, stream>>>(Wq, bq, Wk, bk, Wv, bv, We, Ws, bs, Wnode, ball, Wep);

  int mtiles = (N + 15) / 16;
  k_node<<<(mtiles + 7) / 8, 256, 0, stream>>>(x, Wnode, ball, qb, kb, vb, out, N);

  int etiles = (E + 15) / 16;
  k_edge<<<(etiles + EW - 1) / EW, EW * 32, 0, stream>>>(lu, t, msg, Wt, bt, ei, Wep,
                                                         qb, kb, vb, alpha, wbuf, amax, E);

  k_exp<<<(2 * E + 255) / 256, 256, 0, stream>>>(alpha, amax, denom, ei, E);

  long long tot = (long long)E * HC;
  k_out<<<(unsigned)((tot + 255) / 256), 256, 0, stream>>>(alpha, denom, wbuf, ei, out, E);
}
